// STN_RNN_66013647340231
// MI455X (gfx1250) — matrix operand order repaired and checked by reading, not compiled, not run on hardware
//
#include <hip/hip_runtime.h>
#include <hip/hip_bf16.h>

// ---------------------------------------------------------------------------
// STN_RNN (LSTM cell + state smoothing, k = 0.5) persistent kernel for gfx1250
//   B=32, T=4096, IN=1, H=512, gates N = 4H = 2048
//   16 workgroups x 256 threads (8 wave32s). WG g owns hidden units [32g,32g+32)
//   -> 128 gate columns, W_hh slice (128x512 f16) resident in LDS.
//   Per step:
//     TDM tensor_load_to_lds: h (32x512 f16, 32KB) global -> LDS (padded rows)
//     WMMA f16 GEMM (M=32,N=128,K=512) entirely from LDS
//     LSTM elementwise + STN smoothing; h exchange via double-buffered f16 h
//     in d_ws + one grid-wide atomic barrier per step.
// ---------------------------------------------------------------------------

typedef __attribute__((ext_vector_type(16))) _Float16 v16h;
typedef __attribute__((ext_vector_type(8)))  _Float16 v8h;
typedef __attribute__((ext_vector_type(8)))  float    v8f;
typedef __attribute__((ext_vector_type(4)))  unsigned u32x4;
typedef __attribute__((ext_vector_type(8)))  int      i32x8;
typedef __attribute__((ext_vector_type(4)))  int      i32x4;

#define TSTEPS   4096
#define BATCH    32
#define HID      512
#define NBLOCKS  16
#define UNITS    32            // hidden units per WG  (HID / NBLOCKS)
#define LCOLS    128           // gate columns per WG  (UNITS * 4)
#define WSTRIDE  520           // LDS row stride in halves (512 + 8 pad -> bank skew)

// --- LDS layout (bytes) ---
#define LDS_W        0
#define LDS_W_SZ     (LCOLS * WSTRIDE * 2)                 // 133120
#define LDS_GATES    (LDS_W + LDS_W_SZ)                    // 32 x 128 f32
#define LDS_GATES_SZ (BATCH * LCOLS * 4)                   // 16384
#define LDS_BIAS     (LDS_GATES + LDS_GATES_SZ)
#define LDS_WIH      (LDS_BIAS + LCOLS * 4)
#define LDS_HS       (LDS_WIH + LCOLS * 4)
#define LDS_CS       (LDS_HS + BATCH * UNITS * 4)
#define LDS_HLDS     (LDS_CS + BATCH * UNITS * 4)          // 158720, h stage
#define LDS_HLDS_SZ  (BATCH * WSTRIDE * 2)                 // 33280 (padded rows)
#define LDS_TOTAL    (LDS_HLDS + LDS_HLDS_SZ)              // 192000 (< 320KB WGP LDS)

// --- d_ws layout ---
//  [0,8)      : barrier count / generation
//  [256, ...) : h double buffer: 2 x (32 x 512) f16 = 64 KB

#if defined(__has_builtin)
#if __has_builtin(__builtin_amdgcn_tensor_load_to_lds)
#define HAVE_TDM 1
#endif
#endif

__device__ __forceinline__ void grid_sync(unsigned* cnt, unsigned* gen, unsigned nb) {
    __syncthreads();
    if (threadIdx.x == 0) {
        __threadfence();
        unsigned g = __atomic_load_n(gen, __ATOMIC_RELAXED);
        if (__atomic_fetch_add(cnt, 1u, __ATOMIC_ACQ_REL) == nb - 1u) {
            __atomic_store_n(cnt, 0u, __ATOMIC_RELAXED);
            __atomic_fetch_add(gen, 1u, __ATOMIC_RELEASE);
        } else {
            while (__atomic_load_n(gen, __ATOMIC_ACQUIRE) == g) {
                __builtin_amdgcn_s_sleep(2);
            }
        }
        __threadfence();
    }
    __syncthreads();
}

__device__ __forceinline__ float sigmoidf_dev(float z) {
    return 1.0f / (1.0f + __expf(-z));
}

#define SHUF16(lo, hi) __builtin_shufflevector(lo, hi, 0,1,2,3,4,5,6,7,8,9,10,11,12,13,14,15)

__global__ __launch_bounds__(256, 1)
void stn_lstm_persistent(const float* __restrict__ x,     // (32, 4096, 1)
                         const float* __restrict__ Wih,   // (2048, 1)
                         const float* __restrict__ Whh,   // (2048, 512)
                         const float* __restrict__ bih,   // (2048)
                         const float* __restrict__ bhh,   // (2048)
                         float* __restrict__ out,         // 32*4096*512 + 32*1024
                         unsigned* __restrict__ sync,
                         _Float16* __restrict__ hbuf)     // 2 x (32*512) f16
{
    extern __shared__ char smem[];
    _Float16* Wlds   = (_Float16*)(smem + LDS_W);
    float*    gates  = (float*)(smem + LDS_GATES);
    float*    biasl  = (float*)(smem + LDS_BIAS);
    float*    wihl   = (float*)(smem + LDS_WIH);
    float*    hstate = (float*)(smem + LDS_HS);
    float*    cstate = (float*)(smem + LDS_CS);
    _Float16* hlds   = (_Float16*)(smem + LDS_HLDS);

    const int tid   = threadIdx.x;
    const int wg    = blockIdx.x;        // 0..15
    const int ubase = wg * UNITS;

    // ---- startup: stage W_hh slice f32->f16 into LDS (WMMA-B friendly layout)
    // local column c (0..127): gate q = c>>5, unit j = c&31 -> global row n
    for (int idx = tid; idx < LCOLS * HID; idx += 256) {
        int c = idx >> 9;
        int k = idx & (HID - 1);
        int n = (c >> 5) * HID + ubase + (c & 31);
        Wlds[c * WSTRIDE + k] = (_Float16)Whh[n * HID + k];
    }
    for (int idx = tid; idx < LCOLS * 8; idx += 256) {   // zero the pad halves
        int c = idx >> 3;
        Wlds[c * WSTRIDE + HID + (idx & 7)] = (_Float16)0.0f;
    }
    for (int c = tid; c < LCOLS; c += 256) {
        int n = (c >> 5) * HID + ubase + (c & 31);
        biasl[c] = bih[n] + bhh[n];
        wihl[c]  = Wih[n];
    }
    for (int p = tid; p < BATCH * UNITS; p += 256) {
        hstate[p] = 0.0f;
        cstate[p] = 0.0f;
        int b = p >> 5, j = p & 31;
        hbuf[b * HID + ubase + j] = (_Float16)0.0f;      // zero our slice of ping buf
    }
    grid_sync(sync, sync + 1, gridDim.x);

    // ---- TDM descriptor (D#) pieces: h buffer (32 rows x 512 f16) -> hlds,
    //      with LDS padding: every 1024B (pad_interval=7) insert 16B (pad_amount=3)
    //      so LDS rows land at WSTRIDE = 520 halves.
    const unsigned lds_h_addr = __builtin_amdgcn_groupstaticsize() + (unsigned)LDS_HLDS;
    const unsigned long long ga_ping = (unsigned long long)(uintptr_t)hbuf;
    const unsigned long long ga_pong = ga_ping + (unsigned long long)(BATCH * HID * 2);

    i32x8 tdm_g1;
    tdm_g1[0] = (int)((1u << 16)        // data_size = 1 -> 2 bytes
                    | (1u << 20)        // pad_enable
                    | (7u << 22)        // pad_interval: 256 DWORDs = 1024B
                    | (3u << 25));      // pad_amount:   4 DWORDs  = 16B
    tdm_g1[1] = (int)(512u << 16);      // tensor_dim0[15:0] = 512 (atomic addr 0)
    tdm_g1[2] = (int)(32u << 16);       // tensor_dim0 hi = 0 ; tensor_dim1 = 32
    tdm_g1[3] = (int)(512u << 16);      // tensor_dim1 hi = 0 ; tile_dim0 = 512
    tdm_g1[4] = 32;                     // tile_dim1 = 32 ; tile_dim2 = 0
    tdm_g1[5] = 512;                    // tensor_dim0_stride = 512
    tdm_g1[6] = 0;
    tdm_g1[7] = 0;
    const i32x4 tdm_z4 = {0, 0, 0, 0};
    const i32x8 tdm_z8 = {0, 0, 0, 0, 0, 0, 0, 0};

    // ---- per-lane WMMA addressing constants (wave32)
    const int  wv = tid >> 5;         // wave id 0..7 -> N tile (16 columns)
    const int  l  = tid & 31;
    const int  lr = l & 15;
    const bool hi = (l >= 16);
    // B (W slice) in LDS: lane<16 holds K {0..7,16..23}, lane>=16 K {8..15,24..31}
    const _Float16* wrow = Wlds + (16 * wv + lr) * WSTRIDE + (hi ? 8 : 0);
    // A (h stage) in LDS: lane<16 holds K {0..7,16..23}, lane>=16 K {8..15,24..31}
    const _Float16* arow = hlds + lr * WSTRIDE + (hi ? 8 : 0);
    const int col  = 16 * wv + lr;    // gate column this lane produces
    const int mb   = hi ? 8 : 0;      // D row offset for this half-wave

    for (int t = 0; t < TSTEPS; ++t) {
        const _Float16* hsrc = hbuf + (t & 1) * (BATCH * HID);
        _Float16*       hdst = hbuf + ((t + 1) & 1) * (BATCH * HID);

        // ---- stage h (A matrix) into LDS: one TDM per WG (wave 0 issues)
#ifdef HAVE_TDM
        if (tid < 32) {
            unsigned long long ga = (t & 1) ? ga_pong : ga_ping;
            u32x4 g0;
            g0[0] = 1u;                                    // count = 1 (valid D#)
            g0[1] = lds_h_addr;                            // lds_addr
            g0[2] = (unsigned)ga;                          // global_addr[31:0]
            g0[3] = (unsigned)(ga >> 32) | (2u << 30);     // global_addr[56:32] | type=2
            __builtin_amdgcn_tensor_load_to_lds(g0, tdm_g1, tdm_z4, tdm_z4, tdm_z8, 0);
            __builtin_amdgcn_s_wait_tensorcnt(0);
        }
        __syncthreads();
#else
        for (int p = tid; p < BATCH * HID / 8; p += 256) { // cooperative fallback
            int r = p / (HID / 8), cc = p % (HID / 8);
            *(v8h*)(hlds + r * WSTRIDE + cc * 8) = *(const v8h*)(hsrc + r * HID + cc * 8);
        }
        __syncthreads();
#endif

        v8f acc0 = {};   // batches 0..15
        v8f acc1 = {};   // batches 16..31
#pragma unroll 4
        for (int kc = 0; kc < HID / 32; ++kc) {
            const _Float16* bp = wrow + kc * 32;
            v8h  blo = *(const v8h*)bp;
            v8h  bhi = *(const v8h*)(bp + 16);
            v16h bm  = SHUF16(blo, bhi);

            const _Float16* ap0 = arow + kc * 32;
            v8h  a0lo = *(const v8h*)ap0;
            v8h  a0hi = *(const v8h*)(ap0 + 16);
            v16h am0  = SHUF16(a0lo, a0hi);

            const _Float16* ap1 = ap0 + 16 * WSTRIDE;
            v8h  a1lo = *(const v8h*)ap1;
            v8h  a1hi = *(const v8h*)(ap1 + 16);
            v16h am1  = SHUF16(a1lo, a1hi);

            acc0 = __builtin_amdgcn_wmma_f32_16x16x32_f16(
                       false, am0, false, bm, (short)0, acc0, false, false);
            acc1 = __builtin_amdgcn_wmma_f32_16x16x32_f16(
                       false, am1, false, bm, (short)0, acc1, false, false);
        }

        // scatter gate pre-activations to LDS (D layout: vgpr r -> row r(+8))
#pragma unroll
        for (int r = 0; r < 8; ++r) {
            gates[(mb + r) * LCOLS + col]        = acc0[r];
            gates[(16 + mb + r) * LCOLS + col]   = acc1[r];
        }
        __syncthreads();

        // ---- LSTM cell + STN smoothing (k = 0.5) for our 32 units, all 32 batches
#pragma unroll
        for (int i = 0; i < 4; ++i) {
            int p = tid + 256 * i;            // 0..1023
            int b = p >> 5, j = p & 31;
            float xb = x[b * TSTEPS + t];
            const float* gb = gates + b * LCOLS;
            float ipre = gb[j]        + biasl[j]        + xb * wihl[j];
            float fpre = gb[32 + j]   + biasl[32 + j]   + xb * wihl[32 + j];
            float gpre = gb[64 + j]   + biasl[64 + j]   + xb * wihl[64 + j];
            float opre = gb[96 + j]   + biasl[96 + j]   + xb * wihl[96 + j];
            float ig = sigmoidf_dev(ipre);
            float fg = sigmoidf_dev(fpre);
            float og = sigmoidf_dev(opre);
            float ccell = fg * cstate[p] + ig * tanhf(gpre);
            float hcell = og * tanhf(ccell);
            float hn = 0.5f * (hstate[p] + hcell);
            float cn = 0.5f * (cstate[p] + ccell);
            hstate[p] = hn;
            cstate[p] = cn;
            hdst[b * HID + ubase + j] = (_Float16)hn;
            out[((size_t)b * TSTEPS + t) * HID + ubase + j] = hn;
        }
        grid_sync(sync, sync + 1, gridDim.x);
    }

    // ---- final_state tail: (B, 2H) = [h | c]
    float* fs = out + (size_t)BATCH * TSTEPS * HID;
#pragma unroll
    for (int i = 0; i < 4; ++i) {
        int p = tid + 256 * i;
        int b = p >> 5, j = p & 31;
        fs[b * (2 * HID) + ubase + j]       = hstate[p];
        fs[b * (2 * HID) + HID + ubase + j] = cstate[p];
    }
}

extern "C" void kernel_launch(void* const* d_in, const int* in_sizes, int n_in,
                              void* d_out, int out_size, void* d_ws, size_t ws_size,
                              hipStream_t stream) {
    const float* x   = (const float*)d_in[0];   // (32, 4096, 1)
    const float* Wih = (const float*)d_in[1];   // (2048, 1)
    const float* Whh = (const float*)d_in[2];   // (2048, 512)
    const float* bih = (const float*)d_in[3];   // (2048)
    const float* bhh = (const float*)d_in[4];   // (2048)
    float* out = (float*)d_out;

    unsigned* sync  = (unsigned*)d_ws;
    _Float16* hbuf  = (_Float16*)((char*)d_ws + 256);

    // reset the grid barrier every call (graph-capture safe: memset node)
    (void)hipMemsetAsync(d_ws, 0, 256, stream);

    dim3 grid(NBLOCKS), block(256);
    hipLaunchKernelGGL(stn_lstm_persistent, grid, block, LDS_TOTAL, stream,
                       x, Wih, Whh, bih, bhh, out, sync, hbuf);
}
